// RetinexDecomposer_55422257987787
// MI455X (gfx1250) — matrix operand order repaired and checked by reading, not compiled, not run on hardware
//
#include <hip/hip_runtime.h>

typedef _Float16 h16;
typedef __attribute__((ext_vector_type(16))) _Float16 v16h;
typedef __attribute__((ext_vector_type(8)))  float    v8f;

#define WIDTH   512
#define HEIGHT  512
#define HW      (512 * 512)          // 262144 = 2^18
#define NPLANE  48                   // 16 * 3
#define TOTAL   (NPLANE * HW)        // 12,582,912
#define EPSF    1e-6f
#define CLAMPF  5.0f

// ---------------------------------------------------------------------------
// Weight generation: normalized Gaussian taps for sigma=15 (r=45, T=91) and
// sigma=80 (r=240, T=481). block 0 -> sigma15, block 1 -> sigma80.
// Arrays are chunk-padded (96 / 496 floats) with zeros beyond T.
// ---------------------------------------------------------------------------
__global__ void gen_weights_kernel(float* __restrict__ w15, float* __restrict__ w80) {
    const int which = blockIdx.x;
    const int   R  = which ? 240 : 45;
    const int   T  = 2 * R + 1;
    const int   WP = which ? 496 : 96;
    const float sg = which ? 80.f : 15.f;
    float* w = which ? w80 : w15;
    const int lane = threadIdx.x;            // 32 threads, one wave

    float s = 0.f;
    for (int t = lane; t < WP; t += 32) {
        float g = 0.f;
        if (t < T) {
            float d = (float)(t - R) / sg;
            g = __expf(-0.5f * d * d);
        }
        s += g;
    }
    #pragma unroll
    for (int off = 16; off > 0; off >>= 1) s += __shfl_down(s, off);
    s = __shfl(s, 0);
    const float inv = 1.f / s;
    for (int t = lane; t < WP; t += 32) {
        float g = 0.f;
        if (t < T) {
            float d = (float)(t - R) / sg;
            g = __expf(-0.5f * d * d);
        }
        w[t] = g * inv;
    }
}

__global__ void init_sums_kernel(float* __restrict__ sums) {
    if (threadIdx.x < 96) sums[threadIdx.x] = 0.f;
}

// ---------------------------------------------------------------------------
// Pre-pass: x_pos = softplus(x)+eps, logx = log(x_pos) stored f16.
// Per-plane sum of x_pos via block reduction + one atomic per block.
// ---------------------------------------------------------------------------
__launch_bounds__(256)
__global__ void pre_kernel(const float* __restrict__ x, h16* __restrict__ logx,
                           float* __restrict__ sum_xpos) {
    const int tid = threadIdx.x;
    const int idx = blockIdx.x * 256 + tid;
    float v  = x[idx];
    float sp = (v > 20.f) ? v : log1pf(__expf(v));
    float xp = sp + EPSF;
    logx[idx] = (h16)__logf(xp);

    __shared__ float red[256];
    red[tid] = xp;
    __syncthreads();
    #pragma unroll
    for (int s = 128; s > 0; s >>= 1) {
        if (tid < s) red[tid] += red[tid + s];
        __syncthreads();
    }
    if (tid == 0) atomicAdd(&sum_xpos[blockIdx.x >> 10], red[0]);  // 1024 blocks/plane
}

// ---------------------------------------------------------------------------
// Shared helper: cooperatively build the banded B matrices for every 16-tap
// chunk in the per-lane WMMA layout: s_B[(c*32 + lane)*16 + i] where lane L
// holds column n = L%16, K-block kb = L/16, element i <-> K = kb*8 + i (i<8), 16 + kb*8 + (i-8) (i>=8).
// B[K, n] = w[c*16 + (K - n)] when 0 <= K-n < 16 (banded), else 0.
// ---------------------------------------------------------------------------
template <int NCH, int T>
__device__ __forceinline__ void fill_bands(h16* s_B, const float* __restrict__ w,
                                           int tid) {
    for (int i = tid; i < NCH * 512; i += 256) {
        int c  = i >> 9;
        int r  = i & 511;
        int L  = r >> 4;
        int e  = r & 15;
        int n  = L & 15;
        int kb = L >> 4;
        int tt  = ((e < 8) ? kb * 8 + e : 16 + kb * 8 + (e - 8)) - n;
        int tap = c * 16 + tt;
        float wv = (tt >= 0 && tt < 16 && tap < T) ? w[tap] : 0.f;
        s_B[i] = (h16)wv;
    }
}

// ---------------------------------------------------------------------------
// Horizontal blur via WMMA. Block = 8 waves, one 16-row strip of one plane.
// LDS strip: 16 x PITCH f16 (zero-padded halo materialized), PITCH chosen
// 8-half aligned (b128-friendly) and bank-skewed.
// Each wave computes 64 outputs (4 tiles of 16x16), banded-matmul per chunk.
// ---------------------------------------------------------------------------
template <int R>
__launch_bounds__(256)
__global__ void hblur_kernel(const h16* __restrict__ in, h16* __restrict__ out,
                             const float* __restrict__ w) {
    constexpr int T     = 2 * R + 1;
    constexpr int NCH   = (T + 15) / 16;
    constexpr int RAW   = WIDTH + 2 * R;
    constexpr int PITCH = ((RAW + 7) & ~7) + 8;   // 16B aligned rows, bank skew

    __shared__ alignas(16) h16 s_in[16 * PITCH];
    __shared__ alignas(16) h16 s_B[NCH * 512];

    const int tid   = threadIdx.x;
    const int plane = blockIdx.x >> 5;
    const int y0    = (blockIdx.x & 31) << 4;
    const h16* inp  = in + (size_t)plane * HW;

    for (int i = tid; i < 16 * PITCH; i += 256) {
        int row = i / PITCH, col = i % PITCH;
        int xx  = col - R;
        h16 val = (h16)0.f;
        if (xx >= 0 && xx < WIDTH) val = inp[(y0 + row) * WIDTH + xx];
        s_in[i] = val;
    }
    fill_bands<NCH, T>(s_B, w, tid);
    __syncthreads();

    const int lane = tid & 31;
    const int wave = tid >> 5;
    const int m    = lane & 15;   // A row / B column / D column index
    const int half = lane >> 4;

    union { v16h v; h16 e[16]; } A, B;
    union { v8f  v; float e[8]; } acc;

    for (int xt = 0; xt < 4; ++xt) {
        const int x0 = wave * 64 + xt * 16;
        #pragma unroll
        for (int g = 0; g < 8; ++g) acc.e[g] = 0.f;

        for (int c = 0; c < NCH; ++c) {
            // A: row m, two contiguous 8-half (16B) runs per lane
            const h16* ap = &s_in[m * PITCH + x0 + c * 16 + half * 8];
            #pragma unroll
            for (int i = 0; i < 8; ++i) A.e[i]     = ap[i];
            #pragma unroll
            for (int i = 0; i < 8; ++i) A.e[8 + i] = ap[16 + i];
            // B: precomputed band chunk, contiguous 32B per lane
            const h16* bp = &s_B[(c * 32 + lane) * 16];
            #pragma unroll
            for (int i = 0; i < 16; ++i) B.e[i] = bp[i];

            acc.v = __builtin_amdgcn_wmma_f32_16x16x32_f16(
                false, A.v, false, B.v, (short)0, acc.v, false, false);
        }

        h16* op = out + (size_t)plane * HW;
        const int xo = x0 + m;
        #pragma unroll
        for (int g = 0; g < 8; ++g) {
            int yo = y0 + half * 8 + g;
            op[yo * WIDTH + xo] = (h16)acc.e[g];
        }
    }
}

// ---------------------------------------------------------------------------
// Vertical blur via WMMA. Block = one 16-col strip. LDS stores the strip
// TRANSPOSED: s_in[col * PITCH + row] so per-lane A reads along K (y) are
// contiguous 16B runs, same as the horizontal kernel. Global fill reads stay
// row-major (coalesced). Output: f32 illum (store or accumulate).
// ---------------------------------------------------------------------------
template <int R, bool ACCUM>
__launch_bounds__(256)
__global__ void vblur_kernel(const h16* __restrict__ in, float* __restrict__ illum,
                             const float* __restrict__ w) {
    constexpr int T     = 2 * R + 1;
    constexpr int NCH   = (T + 15) / 16;
    constexpr int RAW   = HEIGHT + 2 * R;
    constexpr int PITCH = ((RAW + 7) & ~7) + 8;

    __shared__ alignas(16) h16 s_in[16 * PITCH];   // [col][row], row 0 <-> y=-R
    __shared__ alignas(16) h16 s_B[NCH * 512];

    const int tid   = threadIdx.x;
    const int plane = blockIdx.x >> 5;
    const int x0g   = (blockIdx.x & 31) << 4;
    const h16* inp  = in + (size_t)plane * HW + x0g;

    for (int i = tid; i < RAW * 16; i += 256) {
        int row = i >> 4, col = i & 15;            // coalesced global read
        int yy  = row - R;
        h16 val = (h16)0.f;
        if (yy >= 0 && yy < HEIGHT) val = inp[yy * WIDTH + col];
        s_in[col * PITCH + row] = val;             // transposed LDS write
    }
    fill_bands<NCH, T>(s_B, w, tid);
    __syncthreads();

    const int lane = tid & 31;
    const int wave = tid >> 5;
    const int m    = lane & 15;   // x offset within strip
    const int half = lane >> 4;

    union { v16h v; h16 e[16]; } A, B;
    union { v8f  v; float e[8]; } acc;

    for (int yt = 0; yt < 4; ++yt) {
        const int y0 = wave * 64 + yt * 16;
        #pragma unroll
        for (int g = 0; g < 8; ++g) acc.e[g] = 0.f;

        for (int c = 0; c < NCH; ++c) {
            const h16* ap = &s_in[m * PITCH + y0 + c * 16 + half * 8];
            #pragma unroll
            for (int i = 0; i < 8; ++i) A.e[i]     = ap[i];
            #pragma unroll
            for (int i = 0; i < 8; ++i) A.e[8 + i] = ap[16 + i];
            const h16* bp = &s_B[(c * 32 + lane) * 16];
            #pragma unroll
            for (int i = 0; i < 16; ++i) B.e[i] = bp[i];

            acc.v = __builtin_amdgcn_wmma_f32_16x16x32_f16(
                false, A.v, false, B.v, (short)0, acc.v, false, false);
        }

        // D[m, n]: n = lane%16 -> y offset; m = half*8+g -> x offset (contiguous)
        float* op = illum + (size_t)plane * HW + (y0 + m) * WIDTH + x0g + half * 8;
        #pragma unroll
        for (int g = 0; g < 8; ++g) {
            if (ACCUM) op[g] += acc.e[g];
            else       op[g]  = acc.e[g];
        }
    }
}

// ---------------------------------------------------------------------------
// Post A: illum (sum of two blurs, stored in out_low) -> low_lin (unscaled)
// and high_lin; accumulate per-plane sum of low_lin.
// ---------------------------------------------------------------------------
__launch_bounds__(256)
__global__ void postA_kernel(const float* __restrict__ x, float* __restrict__ out,
                             float* __restrict__ sums) {
    const int tid = threadIdx.x;
    const int idx = blockIdx.x * 256 + tid;

    float v  = x[idx];
    float sp = (v > 20.f) ? v : log1pf(__expf(v));
    float xp = sp + EPSF;
    float lg = __logf(xp);

    float il   = 0.5f * out[idx];          // illum = (blur15 + blur80) / 2
    float refl = lg - il;
    float ilc  = fminf(fmaxf(il, -CLAMPF), CLAMPF);
    float low  = __expf(ilc);
    float rc   = fminf(fmaxf(refl, -CLAMPF), CLAMPF);
    float high = tanhf(__expf(rc) - 1.f);

    out[idx]         = low;    // unscaled low_lin (scaled in postB)
    out[TOTAL + idx] = high;

    __shared__ float red[256];
    red[tid] = low;
    __syncthreads();
    #pragma unroll
    for (int s = 128; s > 0; s >>= 1) {
        if (tid < s) red[tid] += red[tid + s];
        __syncthreads();
    }
    if (tid == 0) atomicAdd(&sums[48 + (blockIdx.x >> 10)], red[0]);
}

// ---------------------------------------------------------------------------
// Post B: per-plane scale = (mean(x_pos)+eps) / (mean(low_lin)+eps)
// ---------------------------------------------------------------------------
__launch_bounds__(256)
__global__ void postB_kernel(float* __restrict__ out, const float* __restrict__ sums) {
    const int idx   = blockIdx.x * 256 + threadIdx.x;
    const int plane = idx >> 18;
    const float inv_hw = 1.f / (float)HW;
    float num = sums[plane]      * inv_hw + EPSF;
    float den = sums[48 + plane] * inv_hw + EPSF;
    out[idx] *= num / den;
}

// ---------------------------------------------------------------------------
extern "C" void kernel_launch(void* const* d_in, const int* in_sizes, int n_in,
                              void* d_out, int out_size, void* d_ws, size_t ws_size,
                              hipStream_t stream) {
    const float* x = (const float*)d_in[0];
    float* out = (float*)d_out;
    char*  ws  = (char*)d_ws;

    // Workspace layout (d_ws): logx f16 (25 MB) | weights | sums
    h16*   logx = (h16*)ws;
    float* wts  = (float*)(ws + (size_t)TOTAL * sizeof(h16));
    float* w15  = wts;               // 96 floats (padded)
    float* w80  = wts + 96;          // 496 floats (padded)
    float* sums = wts + 96 + 496;    // [0..47]=sum x_pos, [48..95]=sum low_lin

    // d_out reused as scratch before final values are written:
    //   low half  -> f32 illum accumulator
    //   high half -> f16 horizontal-blur intermediate
    float* illum = out;
    h16*   tmpH  = (h16*)(out + TOTAL);

    const int nblk  = TOTAL / 256;   // 49152
    const int bgrid = NPLANE * 32;   // 1536 strips

    gen_weights_kernel<<<2, 32, 0, stream>>>(w15, w80);
    init_sums_kernel<<<1, 96, 0, stream>>>(sums);
    pre_kernel<<<nblk, 256, 0, stream>>>(x, logx, sums);

    hblur_kernel<45><<<bgrid, 256, 0, stream>>>(logx, tmpH, w15);
    vblur_kernel<45, false><<<bgrid, 256, 0, stream>>>(tmpH, illum, w15);
    hblur_kernel<240><<<bgrid, 256, 0, stream>>>(logx, tmpH, w80);
    vblur_kernel<240, true><<<bgrid, 256, 0, stream>>>(tmpH, illum, w80);

    postA_kernel<<<nblk, 256, 0, stream>>>(x, out, sums);
    postB_kernel<<<nblk, 256, 0, stream>>>(out, sums);
}
